// SelfAttention_19937238188149
// MI455X (gfx1250) — matrix operand order repaired and checked by reading, not compiled, not run on hardware
//
#include <hip/hip_runtime.h>

typedef __bf16 bf16;
typedef __attribute__((ext_vector_type(4)))  __bf16 bf16x4;
typedef __attribute__((ext_vector_type(8)))  __bf16 bf16x8;
typedef __attribute__((ext_vector_type(16))) __bf16 bf16x16;
typedef __attribute__((ext_vector_type(8)))  float  f32x8;
typedef __attribute__((ext_vector_type(4)))  int    i32x4;

#define AS1 __attribute__((address_space(1)))
#define AS3 __attribute__((address_space(3)))

#if defined(__gfx1250__) && __has_builtin(__builtin_amdgcn_global_load_async_to_lds_b128)
#define HAVE_ASYNC_LDS 1
#else
#define HAVE_ASYNC_LDS 0
#endif

// ---------------------------------------------------------------------------
// 16B global -> LDS copy: async (ASYNCcnt-tracked) on gfx1250, sync fallback.
// Builtin signature (from clang-22 diagnostic): (i32x4 AS1*, i32x4 AS3*, Ii, Ii)
// ---------------------------------------------------------------------------
__device__ __forceinline__ void cp16_g2l(const bf16* g, bf16* l) {
#if HAVE_ASYNC_LDS
  __builtin_amdgcn_global_load_async_to_lds_b128(
      (AS1 i32x4*)(AS1 void*)(void*)g,
      (AS3 i32x4*)(AS3 void*)(void*)l, 0, 0);
#else
  *(bf16x8*)l = *(const bf16x8*)g;
#endif
}

__device__ __forceinline__ void wait_async_all() {
#if HAVE_ASYNC_LDS
#if __has_builtin(__builtin_amdgcn_s_wait_asynccnt)
  __builtin_amdgcn_s_wait_asynccnt(0);
#else
  asm volatile("s_wait_asynccnt 0" ::: "memory");
#endif
#endif
}

// ---------------------------------------------------------------------------
// f32 -> bf16 conversion (4 elems/thread)
// ---------------------------------------------------------------------------
__global__ __launch_bounds__(256) void cvt_f32_to_bf16(
    const float* __restrict__ src, bf16* __restrict__ dst, int n) {
  int i4 = (blockIdx.x * 256 + threadIdx.x) * 4;
  if (i4 + 3 < n) {
    float4 v = *(const float4*)(src + i4);
    bf16x4 o;
    o[0] = (bf16)v.x; o[1] = (bf16)v.y; o[2] = (bf16)v.z; o[3] = (bf16)v.w;
    *(bf16x4*)(dst + i4) = o;
  }
}

// ---------------------------------------------------------------------------
// LDS tile staging: 128 rows x 32 bf16 (rows padded to 40 elems / 80B to
// spread LDS banks). 512 16B chunks per tile, 2 chunks/thread/tile.
// ---------------------------------------------------------------------------
__device__ __forceinline__ void stage_tiles(const bf16* __restrict__ Ab,
                                            int lda, int mBlk,
                                            const bf16* __restrict__ Bb,
                                            int ldb, int nBlk, int k0,
                                            bf16 (*__restrict__ As)[40],
                                            bf16 (*__restrict__ Bs)[40],
                                            int tid) {
#pragma unroll
  for (int i = 0; i < 2; ++i) {
    const int c = tid + i * 256;
    const int row = c >> 2;
    const int seg = (c & 3) * 8;  // elems; 16B granule
    cp16_g2l(Ab + (size_t)(mBlk + row) * lda + k0 + seg, &As[row][seg]);
    cp16_g2l(Bb + (size_t)(nBlk + row) * ldb + k0 + seg, &Bs[row][seg]);
  }
}

// WMMA A fragment (16x32 bf16 MxK) from an LDS tile.
// lanes 0-15: rows, K 0..7 & 16..23 ; lanes 16-31: same rows, K 8..15 & 24..31
__device__ __forceinline__ bf16x16 lds_a_frag(const bf16* __restrict__ base,
                                              int row, int laneHi) {
  const bf16* p = base + row * 40 + laneHi * 8;
  bf16x8 lo = *(const bf16x8*)(p);
  bf16x8 hi = *(const bf16x8*)(p + 16);
  return __builtin_shufflevector(lo, hi, 0, 1, 2, 3, 4, 5, 6, 7,
                                 8, 9, 10, 11, 12, 13, 14, 15);
}

// WMMA B fragment (32x16 bf16 KxN), tile holds B[N][K]: lane n = column n,
// lanes 0-15 K 0..7 & 16..23, lanes 16-31 K 8..15 & 24..31.
__device__ __forceinline__ bf16x16 lds_b_frag(const bf16* __restrict__ base,
                                              int col, int laneHi) {
  const bf16* p = base + col * 40 + laneHi * 8;
  bf16x8 lo = *(const bf16x8*)(p);
  bf16x8 hi = *(const bf16x8*)(p + 16);
  return __builtin_shufflevector(lo, hi, 0, 1, 2, 3, 4, 5, 6, 7,
                                 8, 9, 10, 11, 12, 13, 14, 15);
}

// ---------------------------------------------------------------------------
// C = A * B^T on bf16 (A: [M,K] lda, B: [N,K] ldb), f32 WMMA accumulation.
// Block tile 128x128, double-buffered LDS K-slabs of 32.
// 8 waves: 4 along M x 2 along N; wave tile 32x64 = 2x4 WMMA accums.
// mode 0: bf16 row-major store (+bias)
// mode 1: bf16 transposed store for Vt[b][n][s] (+bias), batch from M index
// mode 2: f32 row-major store, scaled
// ---------------------------------------------------------------------------
__global__ __launch_bounds__(256) void gemm_wmma_bf16(
    const bf16* __restrict__ A, int lda, long aBatch,
    const bf16* __restrict__ B, int ldb, long bBatch,
    void* __restrict__ Out, int ldc, long oBatch,
    const float* __restrict__ bias, float scale,
    int K, int mode, int seqLen, long vtBatchStride) {
  __shared__ bf16 As[2][128][40];
  __shared__ bf16 Bs[2][128][40];

  const int tid    = threadIdx.x;
  const int lane   = tid & 31;
  const int wave   = tid >> 5;
  const int laneLo = lane & 15;
  const int laneHi = lane >> 4;
  const int waveM  = wave & 3;   // 4 waves along M
  const int waveN  = wave >> 2;  // 2 waves along N
  const int mL = waveM * 32;     // block-local wave origin
  const int nL = waveN * 64;
  const int mBlk = blockIdx.x * 128;
  const int nBlk = blockIdx.y * 128;

  const bf16* Ab = A + (size_t)blockIdx.z * aBatch;
  const bf16* Bb = B + (size_t)blockIdx.z * bBatch;

  f32x8 c[2][4] = {};

  const int nk = K / 32;
  stage_tiles(Ab, lda, mBlk, Bb, ldb, nBlk, 0, As[0], Bs[0], tid);

  int buf = 0;
  for (int ik = 0; ik < nk; ++ik) {
    wait_async_all();   // current buffer's async copies landed (per wave)
    __syncthreads();    // whole block sees them; prior compute reads done
    if (ik + 1 < nk)    // overlap next slab's copies with this slab's math
      stage_tiles(Ab, lda, mBlk, Bb, ldb, nBlk, (ik + 1) * 32, As[buf ^ 1],
                  Bs[buf ^ 1], tid);

    bf16x16 aF[2], bF[4];
#pragma unroll
    for (int tm = 0; tm < 2; ++tm)
      aF[tm] = lds_a_frag(&As[buf][0][0], mL + tm * 16 + laneLo, laneHi);
#pragma unroll
    for (int tn = 0; tn < 4; ++tn)
      bF[tn] = lds_b_frag(&Bs[buf][0][0], nL + tn * 16 + laneLo, laneHi);
#pragma unroll
    for (int tm = 0; tm < 2; ++tm)
#pragma unroll
      for (int tn = 0; tn < 4; ++tn)
        c[tm][tn] = __builtin_amdgcn_wmma_f32_16x16x32_bf16(
            false, aF[tm], false, bF[tn], (short)0, c[tm][tn], false, false);
    buf ^= 1;
  }

  // C/D layout: VGPR r -> row base+laneHi*8+r, col base+laneLo
#pragma unroll
  for (int tm = 0; tm < 2; ++tm) {
#pragma unroll
    for (int tn = 0; tn < 4; ++tn) {
      const int col = nBlk + nL + tn * 16 + laneLo;
      const float bv = bias ? bias[col] : 0.0f;
#pragma unroll
      for (int r = 0; r < 8; ++r) {
        const int row = mBlk + mL + tm * 16 + laneHi * 8 + r;
        const float val = c[tm][tn][r] * scale + bv;
        if (mode == 0) {
          ((bf16*)Out)[(size_t)row * ldc + col] = (bf16)val;
        } else if (mode == 1) {
          const int b = row / seqLen;
          const int s = row - b * seqLen;
          ((bf16*)Out)[(size_t)b * vtBatchStride + (size_t)col * seqLen + s] =
              (bf16)val;
        } else {
          float* O = (float*)Out + (size_t)blockIdx.z * oBatch;
          O[(size_t)row * ldc + col] = val;
        }
      }
    }
  }
}

// ---------------------------------------------------------------------------
// Row softmax over ncol=2048 f32, writing bf16 probs in place over the row's
// own storage (bf16 row stride between rows = 2*ncol -> AV GEMM lda = 4096).
// ---------------------------------------------------------------------------
__global__ __launch_bounds__(256) void softmax_rows(float* __restrict__ S,
                                                    int ncol) {
  float* row = S + (size_t)blockIdx.x * ncol;
  const int t = threadIdx.x;

  float v[8];
  float m = -1e30f;
#pragma unroll
  for (int j = 0; j < 8; ++j) {
    v[j] = row[t + j * 256];
    m = fmaxf(m, v[j]);
  }
#pragma unroll
  for (int off = 16; off > 0; off >>= 1) m = fmaxf(m, __shfl_xor(m, off, 32));

  __shared__ float red[8];
  if ((t & 31) == 0) red[t >> 5] = m;
  __syncthreads();
  float rowMax = red[0];
#pragma unroll
  for (int w = 1; w < 8; ++w) rowMax = fmaxf(rowMax, red[w]);
  __syncthreads();

  float e[8];
  float s = 0.0f;
#pragma unroll
  for (int j = 0; j < 8; ++j) {
    e[j] = __expf(v[j] - rowMax);
    s += e[j];
  }
#pragma unroll
  for (int off = 16; off > 0; off >>= 1) s += __shfl_xor(s, off, 32);
  if ((t & 31) == 0) red[t >> 5] = s;
  __syncthreads();
  float total = 0.0f;
#pragma unroll
  for (int w = 0; w < 8; ++w) total += red[w];
  const float inv = 1.0f / total;

  bf16* prow = (bf16*)row;  // all global reads of this row already done
#pragma unroll
  for (int j = 0; j < 8; ++j) prow[t + j * 256] = (bf16)(e[j] * inv);
}

// ---------------------------------------------------------------------------
// Host launcher
// ---------------------------------------------------------------------------
extern "C" void kernel_launch(void* const* d_in, const int* in_sizes, int n_in,
                              void* d_out, int out_size, void* d_ws,
                              size_t ws_size, hipStream_t stream) {
  const float* x  = (const float*)d_in[0];
  const float* Wq = (const float*)d_in[1];
  const float* bq = (const float*)d_in[2];
  const float* Wk = (const float*)d_in[3];
  const float* bk = (const float*)d_in[4];
  const float* Wv = (const float*)d_in[5];
  const float* bv = (const float*)d_in[6];
  float* out = (float*)d_out;

  const int B = 4, S = 2048, D = 1024;
  const size_t MD = (size_t)B * S * D;  // 8,388,608
  const size_t DD = (size_t)D * D;      // 1,048,576

  bf16* xh  = (bf16*)d_ws;        // MD
  bf16* Wqh = xh + MD;            // DD
  bf16* Wkh = Wqh + DD;           // DD
  bf16* Wvh = Wkh + DD;           // DD
  bf16* Qh  = Wvh + DD;           // MD
  bf16* Kh  = Qh + MD;            // MD
  bf16* Vt  = Kh + MD;            // MD (Vt[b][d][s])
  float* Sc = (float*)(Vt + MD);  // B*S*S f32 (softmax rewrites bf16 in place)

  // 1) downconvert inputs to bf16
  cvt_f32_to_bf16<<<(int)(MD / 1024), 256, 0, stream>>>(x, xh, (int)MD);
  cvt_f32_to_bf16<<<(int)(DD / 1024), 256, 0, stream>>>(Wq, Wqh, (int)DD);
  cvt_f32_to_bf16<<<(int)(DD / 1024), 256, 0, stream>>>(Wk, Wkh, (int)DD);
  cvt_f32_to_bf16<<<(int)(DD / 1024), 256, 0, stream>>>(Wv, Wvh, (int)DD);

  // 2) QKV projections: [8192,1024] x [1024,1024]^T
  dim3 gq((B * S) / 128, D / 128, 1);
  gemm_wmma_bf16<<<gq, 256, 0, stream>>>(xh, D, 0, Wqh, D, 0, Qh, D, 0, bq,
                                         1.0f, D, 0, S, 0);
  gemm_wmma_bf16<<<gq, 256, 0, stream>>>(xh, D, 0, Wkh, D, 0, Kh, D, 0, bk,
                                         1.0f, D, 0, S, 0);
  gemm_wmma_bf16<<<gq, 256, 0, stream>>>(xh, D, 0, Wvh, D, 0, Vt, D, 0, bv,
                                         1.0f, D, 1, S, (long)D * S);

  // 3) scores: S[b] = Q[b] * K[b]^T / sqrt(D)
  dim3 gs(S / 128, S / 128, B);
  gemm_wmma_bf16<<<gs, 256, 0, stream>>>(Qh, D, (long)S * D, Kh, D,
                                         (long)S * D, Sc, S, (long)S * S,
                                         nullptr, 0.03125f, D, 2, S, 0);

  // 4) softmax over keys, bf16 probs in place (row stride 4096 bf16)
  softmax_rows<<<B * S, 256, 0, stream>>>(Sc, S);

  // 5) O[b] = P[b] * V[b]  (B operand = Vt[b]: [N=D rows, K=S])
  dim3 ga(S / 128, D / 128, B);
  gemm_wmma_bf16<<<ga, 256, 0, stream>>>((const bf16*)Sc, 2 * S,
                                         (long)2 * S * S, Vt, S, (long)D * S,
                                         out, D, (long)S * D, nullptr, 1.0f, S,
                                         2, S, 0);
}
